// OnlineSGPModel_81200651698779
// MI455X (gfx1250) — matrix operand order repaired and checked by reading, not compiled, not run on hardware
//
#include <hip/hip_runtime.h>
#include <hip/hip_bf16.h>
#include <hip/hip_fp16.h>

// ---------------- problem sizes (fixed by reference) ----------------
#define B_      16
#define T_      12
#define N_      30000
#define F_      4
#define ORDER_  7
#define OUT_CH_ 126
#define GOUT_   18
#define MLP_H_  64
#define HOR_    12
#define OUT_    4
#define EMB_    32
#define E_      960000
#define NTILES      (N_ / 16)        // 1875 (exact)
#define TILES_TOTAL (B_ * NTILES)    // 30000

typedef __attribute__((ext_vector_type(16))) _Float16 v16h;
typedef __attribute__((ext_vector_type(8)))  float    v8f;

__device__ __forceinline__ float silu_f(float x) {
    return x / (1.0f + __expf(-x));
}

// Inverse of the 16-bit A-fragment K mapping (ISA 7.12.2):
//   forward: e<8 -> K = hi*8 + e ; e>=8 -> K = 16 + hi*8 + (e-8)
//   inverse: hi = (K>>3)&1 ; e = (K&7) | ((K>>4)<<3)
__device__ __forceinline__ int frag_hi(int kl) { return (kl >> 3) & 1; }
__device__ __forceinline__ int frag_e (int kl) { return (kl & 7) | ((kl >> 4) << 3); }

// ---------------- phase-A kernels (sparse pipeline) ----------------

__global__ void deg_kernel(const int* __restrict__ src, const int* __restrict__ dst,
                           const float* __restrict__ w,
                           float* __restrict__ deg_dst, float* __restrict__ deg_src) {
    unsigned e = blockIdx.x * 256u + threadIdx.x;
    if (e >= E_) return;
    float we = w[e];
    atomicAdd(deg_dst + dst[e], we);
    atomicAdd(deg_src + src[e], we);
}

__global__ void norm_kernel(const int* __restrict__ src, const int* __restrict__ dst,
                            const float* __restrict__ w,
                            const float* __restrict__ deg_dst, const float* __restrict__ deg_src,
                            float* __restrict__ wn_fwd, float* __restrict__ wn_bwd) {
    unsigned e = blockIdx.x * 256u + threadIdx.x;
    if (e >= E_) return;
    float we = w[e];
    wn_fwd[e] = we / fmaxf(deg_dst[dst[e]], 1e-8f);
    wn_bwd[e] = we / fmaxf(deg_src[src[e]], 1e-8f);
}

// xt[n][b*4+f] = x[b][T-1][n][f]  (node-major layout, 64 floats per node)
__global__ void xt_kernel(const float* __restrict__ x, float* __restrict__ hop0) {
    unsigned idx = blockIdx.x * 256u + threadIdx.x;
    if (idx >= (unsigned)N_ * 64u) return;
    int n = idx >> 6, c = idx & 63;
    int b = c >> 2, f = c & 3;
    size_t xi = (size_t)b * (T_ * N_ * F_) + (size_t)(T_ - 1) * (N_ * F_) + (size_t)n * F_ + f;
    hop0[idx] = x[xi];
}

// out[scatter[e]][c] += in[gather[e]][c] * wn[e]
__global__ void khop_kernel(const float* __restrict__ in, float* __restrict__ out,
                            const int* __restrict__ gather, const int* __restrict__ scatter,
                            const float* __restrict__ wn) {
    unsigned tid = blockIdx.x * 256u + threadIdx.x;
    if (tid >= (unsigned)E_ * 64u) return;
    int e = tid >> 6, c = tid & 63;
    float v = in[(size_t)gather[e] * 64 + c] * wn[e];
    atomicAdd(&out[(size_t)scatter[e] * 64 + c], v);
}

// ne[n][o] = node_emb[n] @ emb_w[:,o] + emb_b[o], padded to 128 cols
__global__ void ne_kernel(const float* __restrict__ node_emb, const float* __restrict__ emb_w,
                          const float* __restrict__ emb_b, float* __restrict__ ne) {
    unsigned idx = blockIdx.x * 256u + threadIdx.x;
    if (idx >= (unsigned)N_ * 128u) return;
    int n = idx >> 7, o = idx & 127;
    float acc = 0.0f;
    if (o < OUT_CH_) {
        acc = emb_b[o];
        const float* nev = node_emb + (size_t)n * EMB_;
        #pragma unroll 4
        for (int k = 0; k < EMB_; ++k) acc += nev[k] * emb_w[k * OUT_CH_ + o];
    }
    ne[idx] = acc;
}

// Pre-swizzle weights into per-lane WMMA B-fragment order:
//   frag layout [fi][lane][e] halves, 32B contiguous per lane -> v16h loads.
//   B fragment element e of lane l = W[k][col], k = ks*32 + (e<8 ? (l>>4)*8 + e : 16 + (l>>4)*8 + (e-8)),
//   col = ct*16 + (l&15).  mlp: fi = ks*4+ct (16 frags); ro: fi = ks*3+ct (6 frags).
__global__ void pack_kernel(const float* __restrict__ mlp_w, const float* __restrict__ ro_w,
                            _Float16* __restrict__ mlp_frag, _Float16* __restrict__ ro_frag) {
    unsigned idx = blockIdx.x * 256u + threadIdx.x;
    if (idx < 16u * 32u * 16u) {               // 8192 halves
        int e = idx & 15, lane = (idx >> 4) & 31, fi = idx >> 9;
        int ks = fi >> 2, ct = fi & 3;
        int row = lane & 15, hi = lane >> 4;
        int k = ks * 32 + (e < 8 ? hi * 8 + e : 16 + hi * 8 + (e - 8));
        int col = ct * 16 + row;
        float v = (k < OUT_CH_) ? mlp_w[k * MLP_H_ + col] : 0.0f;
        mlp_frag[idx] = (_Float16)v;
    } else if (idx < 8192u + 6u * 32u * 16u) { // 3072 halves
        unsigned j = idx - 8192u;
        int e = j & 15, lane = (j >> 4) & 31, fi = j >> 9;
        int ks = fi / 3, ct = fi % 3;
        int row = lane & 15, hi = lane >> 4;
        int k = ks * 32 + (e < 8 ? hi * 8 + e : 16 + hi * 8 + (e - 8));
        int col = ct * 16 + row;
        ro_frag[j] = (_Float16)ro_w[k * (HOR_ * OUT_) + col];
    }
}

// ---------------- phase-B: fused conv+silu+emb -> WMMA MLP -> WMMA readout ----------------

__global__ __launch_bounds__(32) void fused_kernel(
    const float* __restrict__ hops,       // [7][N][64]
    const float* __restrict__ ne,         // [N][128]
    const float* __restrict__ conv_w,     // [7][4][18]
    const float* __restrict__ conv_b,     // [126]
    const _Float16* __restrict__ mlp_frag,// [16][32][16] pre-swizzled B frags
    const float* __restrict__ mlp_b,      // [64]
    const _Float16* __restrict__ ro_frag, // [6][32][16] pre-swizzled B frags
    const float* __restrict__ ro_b,       // [48]
    float* __restrict__ out)              // [B][HOR][N][OUT]
{
    __shared__ float Xg[16 * ORDER_ * 4];
    // A fragments stored pre-swizzled as halves: [ks][lane][e]
    __shared__ __align__(32) _Float16 HsA[4 * 32 * 16];   // H tile  (16 x 128)
    __shared__ __align__(32) _Float16 H2A[2 * 32 * 16];   // H2 tile (16 x 64)

    const int lane = threadIdx.x;
    const int tile = blockIdx.x;            // 0 .. TILES_TOTAL-1
    const int b    = tile / NTILES;
    const int n0   = (tile - b * NTILES) * 16;

    // ---- stage per-node hop features (7 groups x 4 feats) into LDS ----
    for (int idx = lane; idx < 16 * ORDER_; idx += 32) {
        int i = idx / ORDER_, g = idx - i * ORDER_;
        const float4 v = *(const float4*)(hops + (size_t)g * N_ * 64 + (size_t)(n0 + i) * 64 + b * 4);
        float* p = &Xg[(i * ORDER_ + g) * 4];
        p[0] = v.x; p[1] = v.y; p[2] = v.z; p[3] = v.w;
    }
    __syncthreads();

    // ---- grouped conv (K=4, VALU) + SiLU + emb add -> swizzled f16 A tile ----
    for (int oi = lane; oi < 16 * OUT_CH_; oi += 32) {    // 63 iters, uniform
        int i = oi / OUT_CH_, o = oi - i * OUT_CH_;
        int g = o / GOUT_, j = o - g * GOUT_;
        const float* xg = &Xg[(i * ORDER_ + g) * 4];
        const float* w  = conv_w + (size_t)(g * F_) * GOUT_ + j;
        float acc = conv_b[o];
        acc += xg[0] * w[0 * GOUT_] + xg[1] * w[1 * GOUT_]
             + xg[2] * w[2 * GOUT_] + xg[3] * w[3 * GOUT_];
        float val = silu_f(acc) + ne[(size_t)(n0 + i) * 128 + o];
        int ks = o >> 5, kl = o & 31;
        HsA[(ks * 32 + (i + frag_hi(kl) * 16)) * 16 + frag_e(kl)] = (_Float16)val;
    }
    {   // zero-pad K = 126,127 (32 entries, one per lane)
        int i = lane >> 1, o = OUT_CH_ + (lane & 1);
        int ks = o >> 5, kl = o & 31;
        HsA[(ks * 32 + (i + frag_hi(kl) * 16)) * 16 + frag_e(kl)] = (_Float16)0.0f;
    }
    __syncthreads();

    const int row = lane & 15;
    const int hi  = lane >> 4;
    const v16h* HsAv = (const v16h*)HsA;
    const v16h* mfv  = (const v16h*)mlp_frag;

    // ---- MLP GEMM: (16x128) @ (128x64), f16 in / f32 acc WMMA ----
    v8f acc[4];
    #pragma unroll
    for (int ct = 0; ct < 4; ++ct)
        #pragma unroll
        for (int r = 0; r < 8; ++r) acc[ct][r] = 0.0f;

    #pragma unroll
    for (int ks = 0; ks < 4; ++ks) {
        v16h a = HsAv[ks * 32 + lane];                 // 2x ds_load_b128
        #pragma unroll
        for (int ct = 0; ct < 4; ++ct) {
            v16h bf = mfv[(ks * 4 + ct) * 32 + lane];  // 2x global_load_b128
            acc[ct] = __builtin_amdgcn_wmma_f32_16x16x32_f16(
                false, a, false, bf, (short)0, acc[ct], false, false);
        }
    }

    // bias + SiLU -> swizzled f16 H2 tile (C/D layout: M = r + hi*8, N = lane&15)
    #pragma unroll
    for (int ct = 0; ct < 4; ++ct)
        #pragma unroll
        for (int r = 0; r < 8; ++r) {
            int M = r + hi * 8, col = ct * 16 + row;
            float val = silu_f(acc[ct][r] + mlp_b[col]);
            int ks2 = col >> 5, kl = col & 31;
            H2A[(ks2 * 32 + (M + frag_hi(kl) * 16)) * 16 + frag_e(kl)] = (_Float16)val;
        }
    __syncthreads();

    const v16h* H2Av = (const v16h*)H2A;
    const v16h* rfv  = (const v16h*)ro_frag;

    // ---- readout GEMM: (16x64) @ (64x48), 2 K-steps x 3 col tiles ----
    v8f racc[3];
    #pragma unroll
    for (int ct = 0; ct < 3; ++ct)
        #pragma unroll
        for (int r = 0; r < 8; ++r) racc[ct][r] = 0.0f;

    #pragma unroll
    for (int ks = 0; ks < 2; ++ks) {
        v16h a = H2Av[ks * 32 + lane];
        #pragma unroll
        for (int ct = 0; ct < 3; ++ct) {
            v16h bf = rfv[(ks * 3 + ct) * 32 + lane];
            racc[ct] = __builtin_amdgcn_wmma_f32_16x16x32_f16(
                false, a, false, bf, (short)0, racc[ct], false, false);
        }
    }

    // bias + transpose-store: y[b,n,h,o] -> out[b,h,n,o]
    #pragma unroll
    for (int ct = 0; ct < 3; ++ct)
        #pragma unroll
        for (int r = 0; r < 8; ++r) {
            int M = r + hi * 8, col = ct * 16 + row;
            float y = racc[ct][r] + ro_b[col];
            int h = col >> 2, oo = col & 3;
            out[(((size_t)b * HOR_ + h) * N_ + (n0 + M)) * OUT_ + oo] = y;
        }
}

// ---------------- host launcher ----------------

extern "C" void kernel_launch(void* const* d_in, const int* in_sizes, int n_in,
                              void* d_out, int out_size, void* d_ws, size_t ws_size,
                              hipStream_t stream) {
    (void)in_sizes; (void)n_in; (void)out_size; (void)ws_size;

    const float* x        = (const float*)d_in[0];
    const int*   ei       = (const int*)  d_in[1];
    const float* ew       = (const float*)d_in[2];
    const float* conv_w   = (const float*)d_in[3];
    const float* conv_b   = (const float*)d_in[4];
    const float* node_emb = (const float*)d_in[5];
    const float* emb_w    = (const float*)d_in[6];
    const float* emb_b    = (const float*)d_in[7];
    const float* mlp_w    = (const float*)d_in[8];
    const float* mlp_b    = (const float*)d_in[9];
    const float* ro_w     = (const float*)d_in[10];
    const float* ro_b     = (const float*)d_in[11];
    float* out = (float*)d_out;

    const int* src = ei;
    const int* dst = ei + E_;

    // workspace carve-up (bytes, all 32B aligned)
    char* ws = (char*)d_ws;
    float*    deg_dst  = (float*)(ws + 0);                  // N
    float*    deg_src  = deg_dst + N_;                      // N
    float*    wn_fwd   = (float*)(ws + 240000);             // E
    float*    wn_bwd   = (float*)(ws + 4080000);            // E
    float*    hops     = (float*)(ws + 7920000);            // 7*N*64
    float*    ne       = (float*)(ws + 61680000);           // N*128
    _Float16* mlp_frag = (_Float16*)(ws + 77040000);        // 16*32*16 halves
    _Float16* ro_frag  = (_Float16*)(ws + 77056384);        // 6*32*16 halves

    float* hop[7];
    for (int g = 0; g < 7; ++g) hop[g] = hops + (size_t)g * N_ * 64;

    // zero accumulation buffers (degrees + hops 1..6)
    hipMemsetAsync(deg_dst, 0, 2 * N_ * sizeof(float), stream);
    hipMemsetAsync(hop[1], 0, (size_t)6 * N_ * 64 * sizeof(float), stream);

    deg_kernel <<<E_ / 256, 256, 0, stream>>>(src, dst, ew, deg_dst, deg_src);
    norm_kernel<<<E_ / 256, 256, 0, stream>>>(src, dst, ew, deg_dst, deg_src, wn_fwd, wn_bwd);
    xt_kernel  <<<(N_ * 64) / 256, 256, 0, stream>>>(x, hop[0]);

    const int KG = (E_ * 64) / 256;  // 240000
    // forward hops (gather src -> scatter dst)
    khop_kernel<<<KG, 256, 0, stream>>>(hop[0], hop[1], src, dst, wn_fwd);
    khop_kernel<<<KG, 256, 0, stream>>>(hop[1], hop[2], src, dst, wn_fwd);
    khop_kernel<<<KG, 256, 0, stream>>>(hop[2], hop[3], src, dst, wn_fwd);
    // backward hops (gather dst -> scatter src)
    khop_kernel<<<KG, 256, 0, stream>>>(hop[0], hop[4], dst, src, wn_bwd);
    khop_kernel<<<KG, 256, 0, stream>>>(hop[4], hop[5], dst, src, wn_bwd);
    khop_kernel<<<KG, 256, 0, stream>>>(hop[5], hop[6], dst, src, wn_bwd);

    ne_kernel  <<<(N_ * 128) / 256, 256, 0, stream>>>(node_emb, emb_w, emb_b, ne);
    pack_kernel<<<(16 * 32 * 16 + 6 * 32 * 16) / 256, 256, 0, stream>>>(mlp_w, ro_w, mlp_frag, ro_frag);

    fused_kernel<<<TILES_TOTAL, 32, 0, stream>>>(hops, ne, conv_w, conv_b,
                                                 mlp_frag, mlp_b, ro_frag, ro_b, out);
}
